// Tagger_6227702579418
// MI455X (gfx1250) — matrix operand order repaired and checked by reading, not compiled, not run on hardware
//
#include <hip/hip_runtime.h>
#include <math.h>

// Problem sizes (reference: T=8192, V=50000, E=1024)
#define T_TOK 8192
#define V_VOC 50000
#define V_PAD 50048   // next multiple of 128 (key-chunk size)
#define E_DIM 1024

typedef __attribute__((ext_vector_type(16))) __bf16 bf16x16;
typedef __attribute__((ext_vector_type(8)))  float  f32x8;

union Frag128 { uint4 q[2]; bf16x16 v; };  // one WMMA operand = 2 x b128

// D = A*B + C, bf16 inputs, f32 accumulate (v_wmma_f32_16x16x32_bf16)
__device__ __forceinline__ f32x8 wmma_bf16(bf16x16 a, bf16x16 b, f32x8 c) {
  return __builtin_amdgcn_wmma_f32_16x16x32_bf16(false, a, false, b,
                                                 (short)0, c, false, false);
}

// 16x32 A fragment from a row-major source. Per-lane data (row = lane&15) is
// 16 contiguous k-values starting at 16*hl.
// `row` must point at (row m, k-offset kofs); 16B aligned.
__device__ __forceinline__ bf16x16 load_a_contig(const __bf16* row, int hl) {
  Frag128 r;
  const uint4* p = (const uint4*)row;
  r.q[0] = p[2 * hl];
  r.q[1] = p[2 * hl + 1];
  return r.v;
}

// 32x16 B fragment from a K-major source (k contiguous per column). Per-lane
// data (col = lane&15) is 16 contiguous k-values starting at 16*hl.
// `colp` must point at (k=row0) within this lane's column; 16B aligned.
__device__ __forceinline__ bf16x16 load_b_contig(const __bf16* colp, int hl) {
  Frag128 r;
  const uint4* p = (const uint4*)(colp + 16 * hl);
  r.q[0] = p[0];
  r.q[1] = p[1];
  return r.v;
}

// ---------------------------------------------------------------------------
// fp32 -> bf16 conversion kernels
// ---------------------------------------------------------------------------
__global__ void cvt_bf16_kernel(const float* __restrict__ src,
                                __bf16* __restrict__ dst, size_t n) {
  for (size_t i = blockIdx.x * (size_t)blockDim.x + threadIdx.x; i < n;
       i += (size_t)gridDim.x * blockDim.x)
    dst[i] = (__bf16)src[i];
}

__global__ void cvt_pad_kernel(const float* __restrict__ src,
                               __bf16* __restrict__ dst,
                               size_t srows, size_t drows, size_t cols) {
  const size_t n = drows * cols;
  for (size_t i = blockIdx.x * (size_t)blockDim.x + threadIdx.x; i < n;
       i += (size_t)gridDim.x * blockDim.x) {
    const size_t r = i / cols;
    dst[i] = (r < srows) ? (__bf16)src[i] : (__bf16)0.0f;
  }
}

// Transpose + convert: dst[c * drows + r] = bf16(src[r * cols + c]), rows
// beyond srows zero-filled. 32x32 LDS tiles, 256 threads.
__global__ __launch_bounds__(256) void transpose_cvt_kernel(
    const float* __restrict__ src, __bf16* __restrict__ dst,
    int srows, int drows, int cols) {
  __shared__ float tile[32][33];
  const int rb = blockIdx.x * 32;            // row block (drows domain)
  const int cb = blockIdx.y * 32;            // col block
  const int tx = threadIdx.x & 31, ty = threadIdx.x >> 5;
#pragma unroll
  for (int i = 0; i < 32; i += 8) {
    const int r = rb + ty + i;
    tile[ty + i][tx] = (r < srows) ? src[(size_t)r * cols + (cb + tx)] : 0.0f;
  }
  __syncthreads();
#pragma unroll
  for (int i = 0; i < 32; i += 8) {
    const int c = cb + ty + i;
    dst[(size_t)c * drows + (rb + tx)] = (__bf16)tile[tx][ty + i];
  }
}

// ---------------------------------------------------------------------------
// Q = tokens_bf16 @ weight (weight supplied transposed, K-major). No LDS:
// both operand fragments are contiguous b128 loads. Block = 8 waves covering
// 128 rows x 64 cols; wave w owns rows [w*16, w*16+16), 4 col tiles.
// ---------------------------------------------------------------------------
__global__ __launch_bounds__(256) void qgemm_kernel(
    const __bf16* __restrict__ A, const __bf16* __restrict__ BT,
    __bf16* __restrict__ C) {
  const int tid = threadIdx.x, wave = tid >> 5, lane = tid & 31;
  const int hl = lane >> 4, l16 = lane & 15;
  const int m0 = blockIdx.x * 128 + wave * 16, n0 = blockIdx.y * 64;

  const f32x8 zero8 = {0.f, 0.f, 0.f, 0.f, 0.f, 0.f, 0.f, 0.f};
  f32x8 acc[4];
#pragma unroll
  for (int t = 0; t < 4; ++t) acc[t] = zero8;

  const __bf16* arow = A + (size_t)(m0 + l16) * E_DIM;
#pragma unroll 4
  for (int k0 = 0; k0 < E_DIM; k0 += 32) {
    bf16x16 a = load_a_contig(arow + k0, hl);
#pragma unroll
    for (int ct = 0; ct < 4; ++ct) {
      bf16x16 b =
          load_b_contig(BT + (size_t)(n0 + ct * 16 + l16) * E_DIM + k0, hl);
      acc[ct] = wmma_bf16(a, b, acc[ct]);
    }
  }
#pragma unroll
  for (int ct = 0; ct < 4; ++ct)
#pragma unroll
    for (int r = 0; r < 8; ++r) {
      const int m = m0 + r + 8 * hl;
      const int n = n0 + ct * 16 + l16;
      C[(size_t)m * E_DIM + n] = (__bf16)acc[ct][r];
    }
}

// ---------------------------------------------------------------------------
// Fused flash-softmax "attention" over the vocabulary.
// One WG (8 waves) per 16 query rows; full E=1024 output held in registers
// (8 f32x8 acc tiles per wave). Keys processed in 128-row chunks with online
// softmax; default-embedding key + self-value folded in at the end.
//   QK^T: B fragments from row-major vocab (K = E contiguous)      -> b128
//   P@V : B fragments from transposed vocab (K = keys contiguous)  -> b128
// ---------------------------------------------------------------------------
__global__ __launch_bounds__(256) void attn_kernel(
    const float* __restrict__ tokens, const float* __restrict__ defemb,
    const __bf16* __restrict__ qbf, const __bf16* __restrict__ vbf,
    const __bf16* __restrict__ vbfT, float* __restrict__ out) {
  __shared__ __align__(16) __bf16 Qs[16 * E_DIM];  // 32 KB query tile
  __shared__ float  Ss[16 * 128];                  //  8 KB logits chunk
  __shared__ __align__(16) __bf16 Ps[16 * 128];    //  4 KB softmax numerators
  __shared__ float  red[16 * 16];                  //  1 KB row reductions
  __shared__ float  rowm[16], rowl[16], rowc[16], rowp[16];

  const int tid  = threadIdx.x;
  const int wave = tid >> 5;
  const int lane = tid & 31;
  const int hl   = lane >> 4;
  const int l16  = lane & 15;
  const int qbase = blockIdx.x * 16;

  {  // stage Q tile into LDS (b128 copies)
    const uint4* src = (const uint4*)(qbf + (size_t)qbase * E_DIM);
    uint4* dst = (uint4*)Qs;
    for (int i = tid; i < 16 * E_DIM / 8; i += 256) dst[i] = src[i];
  }
  if (tid < 16) { rowm[tid] = -__builtin_inff(); rowl[tid] = 0.0f; }
  __syncthreads();

  const f32x8 zero8 = {0.f, 0.f, 0.f, 0.f, 0.f, 0.f, 0.f, 0.f};
  f32x8 acc[8];
#pragma unroll
  for (int t = 0; t < 8; ++t) acc[t] = zero8;

  const int g   = tid >> 4;   // softmax row group 0..15 (one per query row)
  const int t16 = tid & 15;

  for (int j0 = 0; j0 < V_PAD; j0 += 128) {
    const int jb    = j0 + wave * 16;  // this wave's 16-key S tile
    const int mykey = jb + l16;        // per-lane key (B-fragment column)

    if (j0 + 128 < V_PAD) {  // global_prefetch_b8 next chunk
      __builtin_prefetch(vbf + (size_t)(mykey + 128) * E_DIM, 0, 1);
      __builtin_prefetch(vbfT + (size_t)(wave * 128 + l16 * 8) * V_PAD + j0 + 128,
                         0, 1);
    }

    // ---- S = Q * K^T over E (32 WMMAs); all operands are b128 loads ----
    f32x8 s = zero8;
    const __bf16* krow = vbf + (size_t)mykey * E_DIM;
    const __bf16* qrow = Qs + (size_t)l16 * E_DIM;
#pragma unroll 4
    for (int kc = 0; kc < E_DIM / 32; ++kc) {
      bf16x16 a = load_a_contig(qrow + kc * 32, hl);
      bf16x16 b = load_b_contig(krow + kc * 32, hl);
      s = wmma_bf16(a, b, s);
    }
#pragma unroll
    for (int r = 0; r < 8; ++r) {  // C layout: row = r + 8*(lane>>4)
      const int m = r + 8 * hl;
      Ss[m * 128 + wave * 16 + l16] =
          (mykey < V_VOC) ? s[r] : -__builtin_inff();
    }
    __syncthreads();

    // ---- online softmax for the 128-key chunk (16 groups x 16 threads) ----
    float vals[8];
    float lm = -__builtin_inff();
#pragma unroll
    for (int i = 0; i < 8; ++i) {
      vals[i] = Ss[g * 128 + t16 * 8 + i];
      lm = fmaxf(lm, vals[i]);
    }
    red[g * 16 + t16] = lm;
    __syncthreads();
    if (t16 == 0) {
      float cm = red[g * 16];
#pragma unroll
      for (int i = 1; i < 16; ++i) cm = fmaxf(cm, red[g * 16 + i]);
      const float mo = rowm[g];
      const float mn = fmaxf(mo, cm);
      rowc[g] = __expf(mo - mn);   // acc correction (exp(-inf)=0 first chunk)
      rowm[g] = mn;
    }
    __syncthreads();
    const float mn = rowm[g];
    float psum = 0.f;
#pragma unroll
    for (int i = 0; i < 8; ++i) {
      const float p = __expf(vals[i] - mn);
      psum += p;
      Ps[g * 128 + t16 * 8 + i] = (__bf16)p;
    }
    red[g * 16 + t16] = psum;
    __syncthreads();
    if (t16 == 0) {
      float s2 = 0.f;
#pragma unroll
      for (int i = 0; i < 16; ++i) s2 += red[g * 16 + i];
      rowl[g] = rowl[g] * rowc[g] + s2;
    }
    __syncthreads();

    // ---- rescale acc, then acc += P @ V (32 WMMAs, b128 operands) ----
    float cr[8];
#pragma unroll
    for (int r = 0; r < 8; ++r) cr[r] = rowc[r + 8 * hl];
#pragma unroll
    for (int ct = 0; ct < 8; ++ct)
#pragma unroll
      for (int r = 0; r < 8; ++r) acc[ct][r] *= cr[r];

    const __bf16* prow = Ps + (size_t)l16 * 128;
#pragma unroll
    for (int kc = 0; kc < 4; ++kc) {
      bf16x16 a = load_a_contig(prow + kc * 32, hl);
#pragma unroll
      for (int ct = 0; ct < 8; ++ct) {
        bf16x16 b = load_b_contig(
            vbfT + (size_t)(wave * 128 + ct * 16 + l16) * V_PAD + j0 + kc * 32,
            hl);
        acc[ct] = wmma_bf16(a, b, acc[ct]);
      }
    }
    __syncthreads();  // protect Ss/Ps for next chunk
  }

  // ---- default-embedding key: logit, softmax update ----
  {
    float sum = 0.f;
    for (int e = t16 * 64; e < t16 * 64 + 64; ++e)
      sum += (float)Qs[g * E_DIM + e] * defemb[e];
    red[g * 16 + t16] = sum;
    __syncthreads();
    if (t16 == 0) {
      float ld = 0.f;
#pragma unroll
      for (int i = 0; i < 16; ++i) ld += red[g * 16 + i];
      const float mo = rowm[g];
      const float mn = fmaxf(mo, ld);
      const float corr = __expf(mo - mn);
      const float p = __expf(ld - mn);
      rowc[g] = corr;
      rowp[g] = p;                       // unnormalized self weight
      rowl[g] = rowl[g] * corr + p;      // final partition function
    }
    __syncthreads();
  }

  // ---- epilogue: out = (acc*corr + p_def * tokens) / l ----
  float cr[8], pd[8], li[8];
#pragma unroll
  for (int r = 0; r < 8; ++r) {
    const int m = r + 8 * hl;
    cr[r] = rowc[m];
    pd[r] = rowp[m];
    li[r] = 1.0f / rowl[m];
  }
#pragma unroll
  for (int ct = 0; ct < 8; ++ct) {
    const int col = wave * 128 + ct * 16 + l16;
#pragma unroll
    for (int r = 0; r < 8; ++r) {
      const int m = r + 8 * hl;
      const size_t idx = (size_t)(qbase + m) * E_DIM + col;
      out[idx] = (acc[ct][r] * cr[r] + pd[r] * tokens[idx]) * li[r];
    }
  }
}

// ---------------------------------------------------------------------------
extern "C" void kernel_launch(void* const* d_in, const int* in_sizes, int n_in,
                              void* d_out, int out_size, void* d_ws,
                              size_t ws_size, hipStream_t stream) {
  (void)in_sizes; (void)n_in; (void)out_size; (void)ws_size;
  const float* tokens = (const float*)d_in[0];
  const float* vocab  = (const float*)d_in[1];
  const float* weight = (const float*)d_in[2];
  const float* defemb = (const float*)d_in[3];
  float* out = (float*)d_out;

  // Workspace (bf16): tokens 16MB | weight^T 2MB | Q 16MB | vocab 100MB |
  //                   vocab^T 100MB
  char* ws = (char*)d_ws;
  __bf16* tokbf = (__bf16*)ws;
  __bf16* wbfT  = tokbf + (size_t)T_TOK * E_DIM;
  __bf16* qbf   = wbfT  + (size_t)E_DIM * E_DIM;
  __bf16* vbf   = qbf   + (size_t)T_TOK * E_DIM;
  __bf16* vbfT  = vbf   + (size_t)V_PAD * E_DIM;

  cvt_bf16_kernel<<<1024, 256, 0, stream>>>(tokens, tokbf,
                                            (size_t)T_TOK * E_DIM);
  transpose_cvt_kernel<<<dim3(E_DIM / 32, E_DIM / 32), 256, 0, stream>>>(
      weight, wbfT, E_DIM, E_DIM, E_DIM);
  cvt_pad_kernel<<<2048, 256, 0, stream>>>(vocab, vbf, V_VOC, V_PAD, E_DIM);
  transpose_cvt_kernel<<<dim3(V_PAD / 32, E_DIM / 32), 256, 0, stream>>>(
      vocab, vbfT, V_VOC, V_PAD, E_DIM);

  qgemm_kernel<<<dim3(T_TOK / 128, E_DIM / 64), 256, 0, stream>>>(tokbf, wbfT,
                                                                  qbf);
  attn_kernel<<<T_TOK / 16, 256, 0, stream>>>(tokens, defemb, qbf, vbf, vbfT,
                                              out);
}
